// NeuralMemory_47571057770919
// MI455X (gfx1250) — matrix operand order repaired and checked by reading, not compiled, not run on hardware
//
#include <hip/hip_runtime.h>
#include <hip/hip_fp16.h>

typedef __attribute__((ext_vector_type(16))) _Float16 v16h;
typedef __attribute__((ext_vector_type(8)))  _Float16 v8h;
typedef __attribute__((ext_vector_type(8)))  float    v8f;
typedef __attribute__((ext_vector_type(4)))  float    v4f;

#define B_ 32
#define S_ 8192
#define D_ 128
#define H_ 256
#define LN_EPS 1e-5f
#define PSTR 40  // B-panel stride in halves: 32 K-values + 8 pad (80B -> 16B aligned, bank stride 20)

// ---------- WMMA fragment helpers (CDNA5 wave32 layouts, CDNA5 ISA) ----------

__device__ inline v16h cat8(v8h lo, v8h hi) {
    return __builtin_shufflevector(lo, hi, 0, 1, 2, 3, 4, 5, 6, 7,
                                   8, 9, 10, 11, 12, 13, 14, 15);
}

// A 16x32 f16: lane holds row M=lane&15; per-lane halves are two contiguous 8-half runs
// (K = k0+khalf..+7 and K = k0+16+khalf..+7, khalf = 8*(lane>=16)) -> two ds_load_b128.
__device__ inline v16h load_a_frag(const _Float16* src, int ld, int m0, int k0, int lane) {
    const _Float16* p = src + (m0 + (lane & 15)) * ld + k0 + ((lane >> 4) << 3);
    return cat8(*(const v8h*)p, *(const v8h*)(p + 16));
}

// B 32x16 f16 from panel layout: panel (kt, n) stores K=kt*32..+31 of column n contiguously.
// Lane needs K = khalf..+7 and K = 16+khalf..+7 (khalf = 8*(lane>=16)) of column n0+(lane&15).
__device__ inline v16h load_b_frag(const _Float16* panels, int N, int k0, int n0, int lane) {
    const _Float16* p = panels + ((k0 >> 5) * N + n0 + (lane & 15)) * PSTR +
                        ((lane >> 4) << 3);
    return cat8(*(const v8h*)p, *(const v8h*)(p + 16));
}

__device__ inline v8f wmma16(v16h a, v16h b, v8f c) {
    return __builtin_amdgcn_wmma_f32_16x16x32_f16(false, a, false, b, (short)0, c,
                                                  false, false);
}

// silu via hardware transcendentals: v_exp_f32 + v_rcp_f32 (no IEEE div sequence)
__device__ inline float silu(float z) {
    return z * __builtin_amdgcn_rcpf(1.f + __expf(-z));
}

// per-lane async copy of 16B global -> LDS (gfx1250 VGLOBAL async path, tracked by ASYNCcnt)
__device__ inline void async_ld16(unsigned lds_off, const void* gaddr) {
    asm volatile("global_load_async_to_lds_b128 %0, %1, off"
                 :: "v"(lds_off), "v"(gaddr) : "memory");
}
__device__ inline void wait_async0() {
    asm volatile("s_wait_asynccnt 0" ::: "memory");
}

__device__ inline v8h cvt8(v4f f0, v4f f1) {
    v8h h;
#pragma unroll
    for (int j = 0; j < 4; ++j) {
        h[j]     = (_Float16)f0[j];
        h[4 + j] = (_Float16)f1[j];
    }
    return h;
}

// ---------- Kernel W: pre-swizzle W1/W2 to f16 column panels in workspace ----------

__global__ __launch_bounds__(256) void kW(const float* __restrict__ W1,
                                          const float* __restrict__ W2,
                                          _Float16* __restrict__ w1p,
                                          _Float16* __restrict__ w2p) {
    int t = blockIdx.x * 256 + threadIdx.x;
    const int PPB1 = (D_ / 32) * H_;  // panels per batch, W1
    const int PPB2 = (H_ / 32) * D_;  // panels per batch, W2
    const float* src;
    _Float16* dst;
    int N;
    if (t < B_ * PPB1) {
        int b = t / PPB1, r = t % PPB1;
        int kt = r / H_, n = r % H_;
        src = W1 + (size_t)b * D_ * H_ + (size_t)(kt * 32) * H_ + n;
        dst = w1p + (size_t)b * PPB1 * PSTR + (size_t)r * PSTR;
        N = H_;
    } else if (t < B_ * (PPB1 + PPB2)) {
        int u = t - B_ * PPB1;
        int b = u / PPB2, r = u % PPB2;
        int kt = r / D_, n = r % D_;
        src = W2 + (size_t)b * H_ * D_ + (size_t)(kt * 32) * D_ + n;
        dst = w2p + (size_t)b * PPB2 * PSTR + (size_t)r * PSTR;
        N = D_;
    } else {
        return;
    }
    _Float16 h[32];
#pragma unroll
    for (int kk = 0; kk < 32; ++kk) h[kk] = (_Float16)src[(size_t)kk * N];
#pragma unroll
    for (int c = 0; c < 4; ++c) *(v8h*)(dst + c * 8) = *(const v8h*)(h + c * 8);
}

// ---------- Kernel A: q16 = silu(x @ Wq^T), sumsq[b][d] += sum_s q^2 ----------

__global__ __launch_bounds__(256) void kA(const float* __restrict__ x,
                                          const float* __restrict__ Wq,
                                          _Float16* __restrict__ q16,
                                          float* __restrict__ sumsq) {
    extern __shared__ char smem[];
    const int LDX = D_ + 8;                              // 136 halves (A-source pad)
    _Float16* xs  = (_Float16*)smem;                     // 128 x LDX; reused as q stage
    _Float16* wqp = xs + 128 * LDX;                      // (D_/32)*D_ panels * PSTR
    float* ssl = (float*)(wqp + (D_ / 32) * D_ * PSTR);  // 128 column partials

    int tid = threadIdx.x, lane = tid & 31, wid = tid >> 5;
    int b = blockIdx.x / (S_ / 128), s0 = (blockIdx.x % (S_ / 128)) * 128;

    if (tid < D_) ssl[tid] = 0.f;

    // stage x tile f32 -> f16 (b128 loads, b128 LDS stores)
    const float* xt = x + ((size_t)b * S_ + s0) * D_;
    for (int i = tid * 8; i < 128 * D_; i += 256 * 8) {
        int r = i >> 7, c = i & 127;
        *(v8h*)(xs + r * LDX + c) =
            cvt8(*(const v4f*)(xt + i), *(const v4f*)(xt + i + 4));
    }
    // stage Wq (transposed) into panels: wqp[(kt*D+n)*PSTR+kk] = Wq[n][kt*32+kk]
    {
        int n = tid >> 1;
        int k0 = (tid & 1) * 64;
        const float* wr = Wq + n * D_ + k0;
#pragma unroll
        for (int kt = 0; kt < 2; ++kt) {
            _Float16* pd = wqp + (((k0 >> 5) + kt) * D_ + n) * PSTR;
#pragma unroll
            for (int c = 0; c < 4; ++c)
                *(v8h*)(pd + c * 8) = cvt8(*(const v4f*)(wr + kt * 32 + c * 8),
                                           *(const v4f*)(wr + kt * 32 + c * 8 + 4));
        }
    }
    __syncthreads();

    int m0 = wid * 16, nl = lane & 15, rofs = (lane >> 4) << 3;
    v16h a0 = load_a_frag(xs, LDX, m0, 0, lane);
    v16h a1 = load_a_frag(xs, LDX, m0, 32, lane);
    v16h a2 = load_a_frag(xs, LDX, m0, 64, lane);
    v16h a3 = load_a_frag(xs, LDX, m0, 96, lane);

#pragma unroll
    for (int nt = 0; nt < 8; ++nt) {
        v8f acc = {};
        acc = wmma16(a0, load_b_frag(wqp, D_, 0, nt * 16, lane), acc);
        acc = wmma16(a1, load_b_frag(wqp, D_, 32, nt * 16, lane), acc);
        acc = wmma16(a2, load_b_frag(wqp, D_, 64, nt * 16, lane), acc);
        acc = wmma16(a3, load_b_frag(wqp, D_, 96, nt * 16, lane), acc);
        int n = nt * 16 + nl;
        float colsq = 0.f;
#pragma unroll
        for (int v = 0; v < 8; ++v) {
            float sil = silu(acc[v]);
            xs[(m0 + v + rofs) * LDX + n] = (_Float16)sil;  // stage into own strip
            colsq += sil * sil;
        }
        colsq += __shfl_xor(colsq, 16, 32);  // combine row-halves
        if (lane < 16) atomicAdd(&ssl[n], colsq);           // ds_add_f32
    }
    __syncthreads();
    if (tid < D_) atomicAdd(&sumsq[b * D_ + tid], ssl[tid]);  // global_atomic_add_f32

    // stream staged q -> global, coalesced b128
    _Float16* qout = q16 + ((size_t)b * S_ + s0) * D_;
    for (int i = tid * 8; i < 128 * D_; i += 256 * 8) {
        int r = i >> 7, c = i & 127;
        *(v8h*)(qout + i) = *(const v8h*)(xs + r * LDX + c);
    }
}

// ---------- Kernel B: rsq = 1 / max(sqrt(sumsq), 1e-12) ----------

__global__ void kB(const float* __restrict__ sumsq, float* __restrict__ rsq) {
    int i = blockIdx.x * blockDim.x + threadIdx.x;
    if (i < B_ * D_) rsq[i] = 1.f / fmaxf(sqrtf(sumsq[i]), 1e-12f);
}

// ---------- Kernel C: per-batch MLP + LayerNorm + residual ----------

__global__ __launch_bounds__(256) void kC(const _Float16* __restrict__ q16,
                                          const float* __restrict__ rsq,
                                          const _Float16* __restrict__ w1p,
                                          const float* __restrict__ b1,
                                          const _Float16* __restrict__ w2p,
                                          const float* __restrict__ b2,
                                          const float* __restrict__ gamma,
                                          const float* __restrict__ beta,
                                          float* __restrict__ out) {
    extern __shared__ char smem[];
    const int LDQ = D_ + 8;   // 136 halves
    const int LDH = H_ + 8;   // 264 halves (528B/row)
    const int LDO = D_ + 4;   // 132 floats (528B/row) -> exact reuse of hs region
    const unsigned QS_OFF = 0;
    const unsigned W1_OFF = 128 * LDQ * 2;                       // 34816
    const unsigned W2_OFF = W1_OFF + (D_ / 32) * H_ * PSTR * 2;  // +81920
    const unsigned HS_OFF = W2_OFF + (H_ / 32) * D_ * PSTR * 2;  // +81920
    _Float16* qs  = (_Float16*)(smem + QS_OFF);
    _Float16* w1s = (_Float16*)(smem + W1_OFF);
    _Float16* w2s = (_Float16*)(smem + W2_OFF);
    _Float16* hs  = (_Float16*)(smem + HS_OFF);
    float*    osc = (float*)(smem + HS_OFF);  // per-wave-strip reuse after ha loaded

    int tid = threadIdx.x, lane = tid & 31, wid = tid >> 5;
    int b = blockIdx.x / (S_ / 128), s0 = (blockIdx.x % (S_ / 128)) * 128;

    // async-copy pre-swizzled f16 weight panels straight into LDS
    const _Float16* w1g = w1p + (size_t)b * (D_ / 32) * H_ * PSTR;
    for (int i = tid; i < (D_ / 32) * H_ * PSTR / 8; i += 256)
        async_ld16(W1_OFF + i * 16, w1g + i * 8);
    const _Float16* w2g = w2p + (size_t)b * (H_ / 32) * D_ * PSTR;
    for (int i = tid; i < (H_ / 32) * D_ * PSTR / 8; i += 256)
        async_ld16(W2_OFF + i * 16, w2g + i * 8);

    // stage normalized q tile (f16 * rsq -> f16)
    const float* rs = rsq + b * D_;
    const _Float16* qt = q16 + ((size_t)b * S_ + s0) * D_;
    for (int i = tid * 8; i < 128 * D_; i += 256 * 8) {
        int r = i >> 7, c = i & 127;
        v8h qv = *(const v8h*)(qt + i);
        v4f r0 = *(const v4f*)(rs + c);
        v4f r1 = *(const v4f*)(rs + c + 4);
        v8h ov;
#pragma unroll
        for (int j = 0; j < 4; ++j) {
            ov[j]     = (_Float16)((float)qv[j] * r0[j]);
            ov[4 + j] = (_Float16)((float)qv[4 + j] * r1[j]);
        }
        *(v8h*)(qs + r * LDQ + c) = ov;
    }
    wait_async0();
    __syncthreads();

    int m0 = wid * 16, nl = lane & 15, rofs = (lane >> 4) << 3;

    // GEMM1: h = silu(q @ W1 + b1); wave owns rows m0..m0+15
    v16h qa0 = load_a_frag(qs, LDQ, m0, 0, lane);
    v16h qa1 = load_a_frag(qs, LDQ, m0, 32, lane);
    v16h qa2 = load_a_frag(qs, LDQ, m0, 64, lane);
    v16h qa3 = load_a_frag(qs, LDQ, m0, 96, lane);
#pragma unroll
    for (int nt = 0; nt < 16; ++nt) {
        v8f acc = {};
        acc = wmma16(qa0, load_b_frag(w1s, H_, 0, nt * 16, lane), acc);
        acc = wmma16(qa1, load_b_frag(w1s, H_, 32, nt * 16, lane), acc);
        acc = wmma16(qa2, load_b_frag(w1s, H_, 64, nt * 16, lane), acc);
        acc = wmma16(qa3, load_b_frag(w1s, H_, 96, nt * 16, lane), acc);
        int n = nt * 16 + nl;
        float bias = b1[b * H_ + n];
#pragma unroll
        for (int v = 0; v < 8; ++v)
            hs[(m0 + v + rofs) * LDH + n] = (_Float16)silu(acc[v] + bias);
    }
    // wave consumes only its own hs strip; LDS ops in-order within wave -> no barrier
    v16h ha[8];
#pragma unroll
    for (int k = 0; k < 8; ++k) ha[k] = load_a_frag(hs, LDH, m0, k * 32, lane);

    float b2v[8], gv[8], bv[8];
#pragma unroll
    for (int nt = 0; nt < 8; ++nt) {
        int n = nt * 16 + nl;
        b2v[nt] = b2[b * D_ + n];
        gv[nt]  = gamma[b * D_ + n];
        bv[nt]  = beta[b * D_ + n];
    }

    // GEMM2: o = h @ W2 + b2 (K=256)
    v8f accAll[8];
#pragma unroll
    for (int nt = 0; nt < 8; ++nt) {
        v8f acc = {};
#pragma unroll
        for (int k = 0; k < 8; ++k)
            acc = wmma16(ha[k], load_b_frag(w2s, D_, k * 32, nt * 16, lane), acc);
#pragma unroll
        for (int v = 0; v < 8; ++v) acc[v] += b2v[nt];
        accAll[nt] = acc;
    }

    // LayerNorm per row; stage ln into osc (f32, own strip)
#pragma unroll
    for (int v = 0; v < 8; ++v) {
        float sum = 0.f, sq = 0.f;
#pragma unroll
        for (int nt = 0; nt < 8; ++nt) {
            float o = accAll[nt][v];
            sum += o;
            sq += o * o;
        }
#pragma unroll
        for (int off = 1; off < 16; off <<= 1) {  // reduce 16 lanes per half
            sum += __shfl_xor(sum, off, 32);
            sq  += __shfl_xor(sq, off, 32);
        }
        float mu   = sum * (1.f / 128.f);
        float var  = sq * (1.f / 128.f) - mu * mu;
        float rstd = rsqrtf(var + LN_EPS);
        int m = m0 + v + rofs;
#pragma unroll
        for (int nt = 0; nt < 8; ++nt) {
            float o = accAll[nt][v];
            osc[m * LDO + nt * 16 + nl] = (o - mu) * rstd * gv[nt] + bv[nt];
        }
    }
    __syncthreads();

    // stream: out = ln + q, coalesced b128 global stores
    float* outt = out + ((size_t)b * S_ + s0) * D_;
    for (int i = tid * 4; i < 128 * D_; i += 256 * 4) {
        int r = i >> 7, c = i & 127;
        v4f o = *(const v4f*)(osc + r * LDO + c);
        const _Float16* qp = qs + r * LDQ + c;
#pragma unroll
        for (int j = 0; j < 4; ++j) o[j] += (float)qp[j];
        *(v4f*)(outt + i) = o;
    }
}

// ---------- host launch ----------

extern "C" void kernel_launch(void* const* d_in, const int* in_sizes, int n_in,
                              void* d_out, int out_size, void* d_ws, size_t ws_size,
                              hipStream_t stream) {
    (void)in_sizes; (void)n_in; (void)out_size; (void)ws_size;
    const float* x     = (const float*)d_in[0];
    const float* Wq    = (const float*)d_in[1];
    const float* W1    = (const float*)d_in[2];
    const float* b1    = (const float*)d_in[3];
    const float* W2    = (const float*)d_in[4];
    const float* b2    = (const float*)d_in[5];
    const float* gamma = (const float*)d_in[6];
    const float* beta  = (const float*)d_in[7];
    float* out = (float*)d_out;

    // workspace: q16 | sumsq | rsq | w1 panels | w2 panels
    char* ws = (char*)d_ws;
    _Float16* q16 = (_Float16*)ws;
    size_t off = (size_t)B_ * S_ * D_ * sizeof(_Float16);
    float* sumsq = (float*)(ws + off);  off += (size_t)B_ * D_ * sizeof(float);
    float* rsq   = (float*)(ws + off);  off += (size_t)B_ * D_ * sizeof(float);
    _Float16* w1p = (_Float16*)(ws + off);
    off += (size_t)B_ * (D_ / 32) * H_ * PSTR * sizeof(_Float16);
    _Float16* w2p = (_Float16*)(ws + off);

    size_t ldsA = (size_t)128 * (D_ + 8) * 2 + (size_t)(D_ / 32) * D_ * PSTR * 2 + D_ * 4;
    size_t ldsC = (size_t)128 * (D_ + 8) * 2 + (size_t)(D_ / 32) * H_ * PSTR * 2 +
                  (size_t)(H_ / 32) * D_ * PSTR * 2 + (size_t)128 * (H_ + 8) * 2;
    (void)hipFuncSetAttribute((const void*)kA,
                              hipFuncAttributeMaxDynamicSharedMemorySize, (int)ldsA);
    (void)hipFuncSetAttribute((const void*)kC,
                              hipFuncAttributeMaxDynamicSharedMemorySize, (int)ldsC);

    (void)hipMemsetAsync(sumsq, 0, (size_t)B_ * D_ * sizeof(float), stream);

    int npanels = B_ * ((D_ / 32) * H_ + (H_ / 32) * D_);
    kW<<<(npanels + 255) / 256, 256, 0, stream>>>(W1, W2, w1p, w2p);

    dim3 grid(B_ * (S_ / 128));
    kA<<<grid, 256, ldsA, stream>>>(x, Wq, q16, sumsq);
    kB<<<(B_ * D_ + 255) / 256, 256, 0, stream>>>(sumsq, rsq);
    kC<<<grid, 256, ldsC, stream>>>(q16, rsq, w1p, b1, w2p, b2, gamma, beta, out);
}
